// BarrierNet_42992622633274
// MI455X (gfx1250) — matrix operand order repaired and checked by reading, not compiled, not run on hardware
//
#include <hip/hip_runtime.h>
#include <cstdint>

typedef __attribute__((ext_vector_type(16))) _Float16 v16h;
typedef __attribute__((ext_vector_type(8)))  float    v8f;
typedef __attribute__((ext_vector_type(4)))  unsigned int v4u;
typedef __attribute__((ext_vector_type(4)))  int      v4i;
typedef __attribute__((ext_vector_type(8)))  int      v8i;

#define NF           8
#define H1           256
#define H2           128
#define WAVES_PER_WG 8
#define ROWS_PER_WG  (WAVES_PER_WG * 16)

// Unified LDS image layout (dword offsets). The prep kernel writes exactly this
// image to workspace; the TDM kernel DMAs it to LDS offset 0 in one shot.
#define OFF_W21    0        // 16384 dwords : W21 f16, WMMA B-fragment order
#define OFF_W22    16384    // 16384 dwords : W22 f16, WMMA B-fragment order
#define OFF_W1     32768    // 1024  dwords : W1 packed f16 (row-major [256][8])
#define OFF_B1     33792    // 256 f32
#define OFF_B2     34048    // 256 f32 (b21 | b22)
#define OFF_W3     34304    // 512 f32 (W31 | W32)
#define OFF_B3     34816    // 4 f32   (b31 | b32)
#define OFF_MEAN   34820    // 8 f32
#define OFF_STD    34828    // 8 f32
#define IMG_DWORDS 34836
#define OFF_HFC    34848    // padded to 64B; 8 waves * 2048 dwords hfc staging
#define SMEM_BYTES ((OFF_HFC + WAVES_PER_WG * 2048) * 4)   // 204928 B < 320 KB

union U16 {
  v16h     h;
  uint32_t u[8];
  _Float16 f[16];
};

__device__ __forceinline__ uint32_t pack2_f16(float a, float b) {
  union { _Float16 f[2]; uint32_t u; } p;
  p.f[0] = (_Float16)a;
  p.f[1] = (_Float16)b;
  return p.u;
}

// ---------------------------------------------------------------------------
// Prep kernel: build the LDS image (f16 conversion + WMMA B-fragment swizzle)
// once in workspace; every WG of the main kernel then DMAs it via TDM.
// ---------------------------------------------------------------------------
extern "C" __global__ __launch_bounds__(256)
void barriernet_prep(const float* __restrict__ meang,
                     const float* __restrict__ stdg,
                     const float* __restrict__ W1g,
                     const float* __restrict__ b1g,
                     const float* __restrict__ W21g,
                     const float* __restrict__ b21g,
                     const float* __restrict__ W22g,
                     const float* __restrict__ b22g,
                     const float* __restrict__ W31g,
                     const float* __restrict__ b31g,
                     const float* __restrict__ W32g,
                     const float* __restrict__ b32g,
                     uint32_t* __restrict__ img)
{
  const int idx = blockIdx.x * blockDim.x + threadIdx.x;
  if (idx >= IMG_DWORDS) return;
  uint32_t val;
  if (idx < OFF_W1) {
    // B-fragment order: dword D = ((t*8)+s)*256 + v*32 + lane
    const int D = idx & 16383;
    const float* W = (idx < OFF_W22) ? W21g : W22g;
    const int t  = D >> 11;
    const int s  = (D >> 8) & 7;
    const int v  = (D >> 5) & 7;
    const int lp = D & 31;
    const int n  = t * 16 + (lp & 15);
    const int k  = s * 32 + (lp >> 4) * 16 + 2 * v;
    val = pack2_f16(W[n * H1 + k], W[n * H1 + k + 1]);
  } else if (idx < OFF_B1) {
    const int e = (idx - OFF_W1) * 2;
    val = pack2_f16(W1g[e], W1g[e + 1]);
  } else {
    float f;
    if      (idx < OFF_B2)   f = b1g[idx - OFF_B1];
    else if (idx < OFF_W3)   { const int e = idx - OFF_B2;  f = (e < H2)  ? b21g[e] : b22g[e - H2]; }
    else if (idx < OFF_B3)   { const int e = idx - OFF_W3;  f = (e < 256) ? W31g[e] : W32g[e - 256]; }
    else if (idx < OFF_MEAN) { const int e = idx - OFF_B3;  f = (e < 2)   ? b31g[e] : b32g[e - 2]; }
    else if (idx < OFF_STD)  f = meang[idx - OFF_MEAN];
    else                     f = stdg[idx - OFF_STD];
    val = __float_as_uint(f);
  }
  img[idx] = val;
}

// ---------------------------------------------------------------------------
// Shared compute body: layer1 WMMA -> hfc A-frag staging -> layers 21/22 WMMA
// with fused layer-3 dot -> wave32 butterfly -> physics + 11-candidate QP.
// ---------------------------------------------------------------------------
__device__ __forceinline__ void barriernet_body(const float* __restrict__ xg,
                                                float* __restrict__ outg,
                                                uint32_t* smemU)
{
  const uint32_t* sW21 = smemU + OFF_W21;
  const uint32_t* sW22 = smemU + OFF_W22;
  const uint32_t* sW1u = smemU + OFF_W1;
  const float*    sB1  = (const float*)(smemU + OFF_B1);
  const float*    sB2  = (const float*)(smemU + OFF_B2);
  const float*    sW3  = (const float*)(smemU + OFF_W3);
  const float*    sB3  = (const float*)(smemU + OFF_B3);
  const float*    sMean= (const float*)(smemU + OFF_MEAN);
  const float*    sStd = (const float*)(smemU + OFF_STD);

  const int tid  = threadIdx.x;
  const int lane = tid & 31;
  const int wave = tid >> 5;
  const int n15  = lane & 15;
  const int half = lane >> 4;
  const int row0 = (blockIdx.x * WAVES_PER_WG + wave) * 16;

  uint32_t* hfcW = smemU + OFF_HFC + wave * 2048;
  _Float16* hfcH = (_Float16*)hfcW;

  // ---- Stage 1: layer1 (K=8 zero-padded to 32) via WMMA ----
  U16 a1;
#pragma unroll
  for (int i = 0; i < 8; ++i) a1.u[i] = 0u;
  if (lane < 16) {
    const float* xr = xg + (row0 + lane) * NF;  // raw x feeds the MLP
#pragma unroll
    for (int i = 0; i < 8; ++i) a1.f[i] = (_Float16)xr[i];
  }

#pragma unroll
  for (int t = 0; t < 16; ++t) {
    U16 b;
#pragma unroll
    for (int i = 0; i < 8; ++i) b.u[i] = 0u;
    if (half == 0) {
#pragma unroll
      for (int v = 0; v < 4; ++v) b.u[v] = sW1u[(t * 16 + n15) * 4 + v]; // K=0..7
    }
    v8f c = {0.f, 0.f, 0.f, 0.f, 0.f, 0.f, 0.f, 0.f};
    c = __builtin_amdgcn_wmma_f32_16x16x32_f16(false, a1.h, false, b.h,
                                               (short)0, c, false, false);
    const float bias = sB1[t * 16 + n15];
    // scatter relu(f16) into layer-2 A-fragment order
    const int K  = t * 16 + n15;
    const int s  = K >> 5;
    const int kk = K & 31;
    const int va = (kk & 15) >> 1;
    const int ha = kk >> 4;
    const int base = (s * 256 + va * 32 + ha * 16) * 2 + (kk & 1);
    const int mb = half * 8;
#pragma unroll
    for (int vc = 0; vc < 8; ++vc) {
      const float val = fmaxf(c[vc] + bias, 0.f);
      hfcH[base + (mb + vc) * 2] = (_Float16)val;
    }
  }

  // ---- Stage 2: layers 21/22 via WMMA + fused layer-3 dot ----
  float a3[2][2][8];
#pragma unroll
  for (int br = 0; br < 2; ++br)
#pragma unroll
    for (int j = 0; j < 2; ++j)
#pragma unroll
      for (int v = 0; v < 8; ++v) a3[br][j][v] = 0.f;

#pragma unroll
  for (int br = 0; br < 2; ++br) {
    const uint32_t* wf = br ? sW22 : sW21;
    v8f acc[8];
#pragma unroll
    for (int t = 0; t < 8; ++t) acc[t] = (v8f){0.f, 0.f, 0.f, 0.f, 0.f, 0.f, 0.f, 0.f};
#pragma unroll
    for (int s = 0; s < 8; ++s) {
      U16 A;
#pragma unroll
      for (int v = 0; v < 8; ++v) A.u[v] = hfcW[s * 256 + v * 32 + lane];
#pragma unroll
      for (int t = 0; t < 8; ++t) {
        U16 Bf;
#pragma unroll
        for (int v = 0; v < 8; ++v) Bf.u[v] = wf[(t * 8 + s) * 256 + v * 32 + lane];
        acc[t] = __builtin_amdgcn_wmma_f32_16x16x32_f16(false, A.h, false, Bf.h,
                                                        (short)0, acc[t], false, false);
      }
    }
#pragma unroll
    for (int t = 0; t < 8; ++t) {
      const int   k    = t * 16 + n15;
      const float bias = sB2[br * H2 + k];
      const float w0   = sW3[br * 256 + k];
      const float w1   = sW3[br * 256 + H2 + k];
#pragma unroll
      for (int v = 0; v < 8; ++v) {
        const float val = fmaxf(acc[t][v] + bias, 0.f);
        a3[br][0][v] += val * w0;
        a3[br][1][v] += val * w1;
      }
    }
  }

  // butterfly reduce across each 16-lane half (C layout halves hold M=v / M=v+8)
#pragma unroll
  for (int mask = 1; mask <= 8; mask <<= 1) {
#pragma unroll
    for (int br = 0; br < 2; ++br)
#pragma unroll
      for (int j = 0; j < 2; ++j)
#pragma unroll
        for (int v = 0; v < 8; ++v)
          a3[br][j][v] += __shfl_xor(a3[br][j][v], mask, 32);
  }

  // route sample m's 4 reduced values to lane m
  const int src = (lane & 8) ? (lane + 8) : lane;
  float x31v0 = 0.f, x31v1 = 0.f, x32p0 = 0.f, x32p1 = 0.f;
#pragma unroll
  for (int v = 0; v < 8; ++v) {
    const float t00 = __shfl(a3[0][0][v], src, 32);
    const float t01 = __shfl(a3[0][1][v], src, 32);
    const float t10 = __shfl(a3[1][0][v], src, 32);
    const float t11 = __shfl(a3[1][1][v], src, 32);
    if ((lane & 7) == v) { x31v0 = t00; x31v1 = t01; x32p0 = t10; x32p1 = t11; }
  }

  // ---- Stage 3: physics + QP, one sample per lane (lanes 0-15) ----
  if (lane < 16) {
    const float* xr = xg + (row0 + lane) * NF;
    float x0[8];
#pragma unroll
    for (int i = 0; i < 8; ++i) x0[i] = xr[i] * sStd[i] + sMean[i];

    const float p0 = x31v0 + sB3[0];
    const float p1 = x31v1 + sB3[1];
    const float c1 = 4.f / (1.f + expf(-(x32p0 + sB3[2])));
    const float c2 = 4.f / (1.f + expf(-(x32p1 + sB3[3])));

    const float px = x0[0], py = x0[1], th = x0[2], vv = x0[3];
    const float st = sinf(th), ct = cosf(th);

    const float OX[4] = {8.f, -9.f, 10.f, 0.f};
    const float OY[4] = {-8.f, 7.f, 10.f, 0.f};
    const float RT[4] = {1.6f, 1.6f, 2.1f, 1.1f};
    float G0[4], G1[4], hh[4];
#pragma unroll
    for (int i = 0; i < 4; ++i) {
      const float ox = (i < 3) ? OX[i] : x0[4];
      const float oy = (i < 3) ? OY[i] : x0[5];
      const float dx = px - ox, dy = py - oy;
      const float bar  = dx * dx + dy * dy - RT[i] * RT[i];
      const float bdot = 2.f * dx * vv * ct + 2.f * dy * vv * st;
      const float L1   = -2.f * dx * vv * st + 2.f * dy * vv * ct;
      const float L2   =  2.f * dx * ct + 2.f * dy * st;
      G0[i] = -L1;
      G1[i] = -L2;
      hh[i] = 2.f * vv * vv + (c1 + c2) * bdot + (c1 * c2) * bar;
    }

    const float tol = 1e-6f, eps = 1e-9f;
    float bu0 = -p0, bu1 = -p1;
    float bestObj = 3.0e38f;

    auto tryCand = [&](float u0c, float u1c, bool ok) {
      bool prim = ok;
#pragma unroll
      for (int i = 0; i < 4; ++i)
        prim = prim && (G0[i] * u0c + G1[i] * u1c <= hh[i] + tol);
      const float o = prim ? (0.5f * (u0c * u0c + u1c * u1c) + p0 * u0c + p1 * u1c)
                           : 1.0e30f;
      if (o < bestObj) { bestObj = o; bu0 = u0c; bu1 = u1c; }
    };

    tryCand(-p0, -p1, true);
#pragma unroll
    for (int i = 0; i < 4; ++i) {
      const float gg  = G0[i] * G0[i] + G1[i] * G1[i];
      const float lam = (G0[i] * (-p0) + G1[i] * (-p1) - hh[i]) / (gg + eps);
      tryCand(-p0 - lam * G0[i], -p1 - lam * G1[i], lam >= -tol);
    }
    const int II[6] = {0, 0, 0, 1, 1, 2};
    const int JJ[6] = {1, 2, 3, 2, 3, 3};
#pragma unroll
    for (int q = 0; q < 6; ++q) {
      const int i = II[q], j = JJ[q];
      const float det = G0[i] * G1[j] - G1[i] * G0[j];
      const bool  ok  = fabsf(det) > eps;
      const float sd  = ok ? det : 1.0f;
      const float u0c = (hh[i] * G1[j] - hh[j] * G1[i]) / sd;
      const float u1c = (G0[i] * hh[j] - G0[j] * hh[i]) / sd;
      const float r0  = -(u0c + p0), r1 = -(u1c + p1);
      const float li  = (r0 * G1[j] - r1 * G0[j]) / sd;
      const float lj  = (G0[i] * r1 - G1[i] * r0) / sd;
      tryCand(u0c, u1c, ok && (li >= -tol) && (lj >= -tol));
    }

    float* o = outg + (row0 + lane) * 2;
    o[0] = bu0;
    o[1] = bu1;
  }
}

// ---------------------------------------------------------------------------
// Main kernel, TDM variant: one tensor_load_to_lds DMA pulls the whole
// pre-swizzled 139 KB weight image (L2-resident) into LDS.
// ---------------------------------------------------------------------------
extern "C" __global__ __launch_bounds__(256)
void barriernet_fused_tdm(const float* __restrict__ xg,
                          const uint32_t* __restrict__ img,
                          float* __restrict__ outg)
{
  extern __shared__ unsigned char smem_raw[];
  uint32_t* smemU = (uint32_t*)smem_raw;

  if ((threadIdx.x >> 5) == 0) {
    // Build Tensor DMA Descriptor (D#) per CDNA5 ISA §8: 1 valid descriptor,
    // 2D tensor, data_size=4B, one row of IMG_DWORDS elements, lds_addr=0
    // (dynamic LDS starts at offset 0; no static LDS in this kernel).
    const unsigned long long ga = (unsigned long long)img;
    v4u g0;
    g0[0] = 1u;                                               // count=1, user mode
    g0[1] = 0u;                                               // lds_addr = 0
    g0[2] = (unsigned int)(ga & 0xFFFFFFFFu);                 // global_addr[31:0]
    g0[3] = (unsigned int)((ga >> 32) & 0x01FFFFFFu) | (2u << 30); // addr[56:32] | type=2

    v8i g1;
    g1[0] = (int)(2u << 16);                                  // mask=0 | data_size=4B
    g1[1] = (int)((IMG_DWORDS & 0xFFFFu) << 16);              // abar=0 | tensor_dim0.lo
    g1[2] = (int)((IMG_DWORDS >> 16) | (1u << 16));           // tensor_dim0.hi | tensor_dim1.lo=1
    g1[3] = (int)((IMG_DWORDS & 0xFFFFu) << 16);              // tensor_dim1.hi=0 | tile_dim0
    g1[4] = 1;                                                // tile_dim1=1 | tile_dim2=0
    g1[5] = (int)IMG_DWORDS;                                  // tensor_dim0_stride[31:0]
    g1[6] = 0;                                                // stride0.hi | stride1.lo
    g1[7] = 0;                                                // stride1.hi

    const v4i gz = {0, 0, 0, 0};
#if __clang_major__ >= 23
    const v8i gz8 = {0, 0, 0, 0, 0, 0, 0, 0};
    __builtin_amdgcn_tensor_load_to_lds(g0, g1, gz, gz, gz8, 0);
#else
    __builtin_amdgcn_tensor_load_to_lds(g0, g1, gz, gz, 0);
#endif
    __builtin_amdgcn_s_wait_tensorcnt(0);
  }
  __syncthreads();

  barriernet_body(xg, outg, smemU);
}

// ---------------------------------------------------------------------------
// Fallback variant: in-kernel staging (no workspace dependency).
// ---------------------------------------------------------------------------
extern "C" __global__ __launch_bounds__(256)
void barriernet_fused_inline(const float* __restrict__ xg,
                             const float* __restrict__ meang,
                             const float* __restrict__ stdg,
                             const float* __restrict__ W1g,
                             const float* __restrict__ b1g,
                             const float* __restrict__ W21g,
                             const float* __restrict__ b21g,
                             const float* __restrict__ W22g,
                             const float* __restrict__ b22g,
                             const float* __restrict__ W31g,
                             const float* __restrict__ b31g,
                             const float* __restrict__ W32g,
                             const float* __restrict__ b32g,
                             float* __restrict__ outg)
{
  extern __shared__ unsigned char smem_raw[];
  uint32_t* smemU = (uint32_t*)smem_raw;

  uint32_t* sW21 = smemU + OFF_W21;
  uint32_t* sW22 = smemU + OFF_W22;
  _Float16* sW1h = (_Float16*)(smemU + OFF_W1);
  float*    sB1  = (float*)(smemU + OFF_B1);
  float*    sB2  = (float*)(smemU + OFF_B2);
  float*    sW3  = (float*)(smemU + OFF_W3);
  float*    sB3  = (float*)(smemU + OFF_B3);
  float*    sMean= (float*)(smemU + OFF_MEAN);
  float*    sStd = (float*)(smemU + OFF_STD);

  const int tid = threadIdx.x;
  for (int e = tid; e < H1 * NF; e += 256) sW1h[e] = (_Float16)W1g[e];
  sB1[tid] = b1g[tid];
  if (tid < H2) { sB2[tid] = b21g[tid]; sB2[H2 + tid] = b22g[tid]; }
  sW3[tid] = W31g[tid];
  sW3[256 + tid] = W32g[tid];
  if (tid < 2) { sB3[tid] = b31g[tid]; sB3[2 + tid] = b32g[tid]; }
  if (tid < 8) { sMean[tid] = meang[tid]; sStd[tid] = stdg[tid]; }

  const float2* W21g2 = (const float2*)W21g;
  const float2* W22g2 = (const float2*)W22g;
  for (int D = tid; D < 16384; D += 256) {
    const int t  = D >> 11;
    const int s  = (D >> 8) & 7;
    const int v  = (D >> 5) & 7;
    const int lp = D & 31;
    const int n  = t * 16 + (lp & 15);
    const int k  = s * 32 + (lp >> 4) * 16 + 2 * v;
    const int i2 = (n * H1 + k) >> 1;
    float2 w = W21g2[i2];
    sW21[D] = pack2_f16(w.x, w.y);
    w = W22g2[i2];
    sW22[D] = pack2_f16(w.x, w.y);
  }
  __syncthreads();

  barriernet_body(xg, outg, smemU);
}

// ---------------------------------------------------------------------------
extern "C" void kernel_launch(void* const* d_in, const int* in_sizes, int n_in,
                              void* d_out, int out_size, void* d_ws, size_t ws_size,
                              hipStream_t stream) {
  const float* x    = (const float*)d_in[0];
  const float* mean = (const float*)d_in[1];
  const float* std_ = (const float*)d_in[2];
  const float* W1   = (const float*)d_in[3];
  const float* b1   = (const float*)d_in[4];
  const float* W21  = (const float*)d_in[5];
  const float* b21  = (const float*)d_in[6];
  const float* W22  = (const float*)d_in[7];
  const float* b22  = (const float*)d_in[8];
  const float* W31  = (const float*)d_in[9];
  const float* b31  = (const float*)d_in[10];
  const float* W32  = (const float*)d_in[11];
  const float* b32  = (const float*)d_in[12];
  // d_in[13] = sgn (forward pass only)
  float* out = (float*)d_out;

  const int rows   = in_sizes[0] / NF;          // 262144
  const int blocks = rows / ROWS_PER_WG;        // 2048

  if (ws_size >= (size_t)IMG_DWORDS * 4) {
    uint32_t* img = (uint32_t*)d_ws;
    barriernet_prep<<<(IMG_DWORDS + 255) / 256, 256, 0, stream>>>(
        mean, std_, W1, b1, W21, b21, W22, b22, W31, b31, W32, b32, img);
    barriernet_fused_tdm<<<blocks, 256, SMEM_BYTES, stream>>>(x, img, out);
  } else {
    barriernet_fused_inline<<<blocks, 256, SMEM_BYTES, stream>>>(
        x, mean, std_, W1, b1, W21, b21, W22, b22, W31, b31, W32, b32, out);
  }
}
